// CGFormerEncoder_24464133718318
// MI455X (gfx1250) — matrix operand order repaired and checked by reading, not compiled, not run on hardware
//
#include <hip/hip_runtime.h>
#include <hip/hip_fp16.h>

// ---------------------------------------------------------------------------
// CGFormer encoder for MI455X (gfx1250, wave32).
//  - Dense GEMMs on v_wmma_f32_16x16x32_f16, f32 accumulate.
//  - Weights pre-swizzled to f16 in WMMA lane order: B chunk = one contiguous
//    32B per-lane read (2x global_load_b128), fully coalesced.
//  - Each wave computes a 16x64 C block (4 accumulators); A/B streams are
//    software-pipelined one K-chunk ahead so WMMAs don't wait on their loads.
//  - Conv gather/concat fused into A fetch (per-chunk segment base table).
//  - Masked dense attention computed sparsely over <=12 distinct neighbors.
//  - BatchNorm: deterministic two-stage column reduction (no FP atomics).
// ---------------------------------------------------------------------------

#define NN    4096
#define MM    12
#define ORIGF 92
#define ORIGP 96      // ORIG padded to multiple of 32
#define DD    128
#define NBRF  64
#define NHEAD 4
#define EPSB  1e-5f

typedef __attribute__((ext_vector_type(16))) _Float16 v16h;
typedef __attribute__((ext_vector_type(8)))  float    v8f;

__device__ __forceinline__ float sigmoidf_(float x) { return 1.f / (1.f + expf(-x)); }
__device__ __forceinline__ float softplusf_(float x) { return (x > 20.f) ? x : log1pf(expf(x)); }

// Load one 16x32 A chunk (f32 -> f16) at WMMA layout positions.
__device__ __forceinline__ v16h load_a_chunk(const float* __restrict__ base, int aSel) {
  v16h a;
#pragma unroll
  for (int vv = 0; vv < 8; ++vv) {
    int ka = aSel + (vv & 3) * 2 + ((vv >> 2) << 4);
    float2 av = *(const float2*)(base + ka);
    a[2 * vv]     = (_Float16)av.x;
    a[2 * vv + 1] = (_Float16)av.y;
  }
  return a;
}

// Pre-swizzled B: chunk (tn,kc) stored as 32 lanes x 16 f16 contiguously.
__device__ __forceinline__ v16h load_b_swz(const _Float16* __restrict__ Bs,
                                           int chunkIdx, int lane) {
  return *(const v16h*)(Bs + (((size_t)chunkIdx * 32 + lane) << 4));
}

// Swizzle f32 row-major B[K,NC] into WMMA-lane-order f16.
// dst index i = ((tn*KC + kc)*32 + lane)*16 + e ; e -> (vv=e>>1, t=e&1)
__global__ void k_swz_b(const float* __restrict__ B, _Float16* __restrict__ Bs,
                        int K, int NC) {
  int i = blockIdx.x * blockDim.x + threadIdx.x;
  if (i >= K * NC) return;
  int e        = i & 15;
  int lane     = (i >> 4) & 31;
  int chunkIdx = i >> 9;
  int KC = K >> 5;
  int kc = chunkIdx % KC;
  int tn = chunkIdx / KC;
  int k   = kc * 32 + ((lane >> 4) << 3) + (e & 7) + ((e >> 3) << 4);
  int col = tn * 16 + (lane & 15);
  Bs[i] = (_Float16)B[k * NC + col];
}

// ------------------------- generic WMMA GEMM -------------------------------
// C[Mr,128] = act( A[Mr,K] @ B[K,128] + bias (+ add) ); B pre-swizzled f16.
// One wave computes 16 rows x 64 cols, pipelined one K-chunk ahead.
// ACT: 0=none, 1=relu, 2=silu
template <int K, int ACT, bool HAS_ADD>
__global__ void k_gemm_f16(const float* __restrict__ A, const _Float16* __restrict__ Bs,
                           const float* __restrict__ bias, const float* __restrict__ add,
                           float* __restrict__ C, int Mr) {
  constexpr int NC = 128;
  constexpr int KC = K / 32;
  constexpr int NG = NC / 64;                    // groups of 4 col-tiles
  const int lane = threadIdx.x & 31;
  const int wid  = blockIdx.x * (blockDim.x >> 5) + (threadIdx.x >> 5);
  const int total = (Mr >> 4) * NG;
  if (wid >= total) return;                      // wave-uniform exit
  const int tm = wid / NG, tg = wid % NG;

  const int aSel = (lane >> 4) * 8;
  const int rowA = tm * 16 + (lane & 15);
  const float* Arow = A + (size_t)rowA * K;

  const v8f zero = {0.f, 0.f, 0.f, 0.f, 0.f, 0.f, 0.f, 0.f};
  v8f acc[4] = {zero, zero, zero, zero};

  // prologue: chunk 0
  v16h aCur = load_a_chunk(Arow, aSel);
  v16h bCur[4];
#pragma unroll
  for (int u = 0; u < 4; ++u) bCur[u] = load_b_swz(Bs, (tg * 4 + u) * KC, lane);

#pragma unroll
  for (int kc = 0; kc < KC; ++kc) {
    const int kn = (kc + 1 < KC) ? kc + 1 : kc;  // clamped prefetch index
    v16h aNext = load_a_chunk(Arow + kn * 32, aSel);
    v16h bNext[4];
#pragma unroll
    for (int u = 0; u < 4; ++u) bNext[u] = load_b_swz(Bs, (tg * 4 + u) * KC + kn, lane);
#pragma unroll
    for (int u = 0; u < 4; ++u)
      acc[u] = __builtin_amdgcn_wmma_f32_16x16x32_f16(false, aCur, false, bCur[u],
                                                      (short)0, acc[u], false, false);
    aCur = aNext;
#pragma unroll
    for (int u = 0; u < 4; ++u) bCur[u] = bNext[u];
  }

  const int rbase = tm * 16 + (lane >> 4) * 8;
#pragma unroll
  for (int u = 0; u < 4; ++u) {
    int col = (tg * 4 + u) * 16 + (lane & 15);
    float bcol = bias[col];
#pragma unroll
    for (int vv = 0; vv < 8; ++vv) {
      int row = rbase + vv;
      float val = acc[u][vv] + bcol;
      if (HAS_ADD) val += add[row * NC + col];
      if (ACT == 1)      val = fmaxf(val, 0.f);
      else if (ACT == 2) val = val * sigmoidf_(val);
      C[row * NC + col] = val;
    }
  }
}

template <int K, int ACT, bool HAS_ADD>
static void launch_gemm(const float* A, const _Float16* Bs, const float* bias,
                        const float* add, float* C, int Mr, hipStream_t stream) {
  int waves = (Mr / 16) * 2;                     // NG = 2 for NC = 128
  k_gemm_f16<K, ACT, HAS_ADD><<<(waves + 3) / 4, 128, 0, stream>>>(A, Bs, bias, add, C, Mr);
}

// ------------- conv GEMM with fused gather/concat A operand ----------------
// g[N*M, 256] = concat(x[n], x[nbr[n,m]], nbr_fea[n,m]) @ fc_W + fc_b
// One wave computes 16 rows x 64 cols (tg in 0..3), pipelined one chunk ahead.
__global__ void k_conv_gemm(const float* __restrict__ x,
                            const float* __restrict__ nbrfea,
                            const int*   __restrict__ nbridx,
                            const _Float16* __restrict__ Bs,
                            const float* __restrict__ fcb,
                            float* __restrict__ g) {
  constexpr int KC = 10;                         // K = 320
  const int lane = threadIdx.x & 31;
  const int wid  = blockIdx.x * (blockDim.x >> 5) + (threadIdx.x >> 5);
  const int tm = wid >> 2, tg = wid & 3;

  const int row = tm * 16 + (lane & 15);         // row in [0, N*M)
  const int n   = row / MM;
  const int j   = nbridx[row];
  const float* px  = x + n * DD;
  const float* pnb = x + j * DD;
  const float* pf  = nbrfea + row * NBRF;
  const int aSel = (lane >> 4) * 8;

  // per-chunk A segment base table (concat boundaries are chunk-aligned)
  const float* abase[KC] = {px,  px  + 32, px  + 64, px  + 96,
                            pnb, pnb + 32, pnb + 64, pnb + 96,
                            pf,  pf  + 32};

  const v8f zero = {0.f, 0.f, 0.f, 0.f, 0.f, 0.f, 0.f, 0.f};
  v8f acc[4] = {zero, zero, zero, zero};

  v16h aCur = load_a_chunk(abase[0], aSel);
  v16h bCur[4];
#pragma unroll
  for (int u = 0; u < 4; ++u) bCur[u] = load_b_swz(Bs, (tg * 4 + u) * KC, lane);

#pragma unroll
  for (int kc = 0; kc < KC; ++kc) {
    const int kn = (kc + 1 < KC) ? kc + 1 : kc;
    v16h aNext = load_a_chunk(abase[kn], aSel);
    v16h bNext[4];
#pragma unroll
    for (int u = 0; u < 4; ++u) bNext[u] = load_b_swz(Bs, (tg * 4 + u) * KC + kn, lane);
#pragma unroll
    for (int u = 0; u < 4; ++u)
      acc[u] = __builtin_amdgcn_wmma_f32_16x16x32_f16(false, aCur, false, bCur[u],
                                                      (short)0, acc[u], false, false);
    aCur = aNext;
#pragma unroll
    for (int u = 0; u < 4; ++u) bCur[u] = bNext[u];
  }

  const int rbase = tm * 16 + (lane >> 4) * 8;
#pragma unroll
  for (int u = 0; u < 4; ++u) {
    int col = (tg * 4 + u) * 16 + (lane & 15);
    float bcol = fcb[col];
#pragma unroll
    for (int vv = 0; vv < 8; ++vv)
      g[(rbase + vv) * 256 + col] = acc[u][vv] + bcol;
  }
}

// -------------------- pad atom_fea [N,92] -> [N,96] ------------------------
__global__ void k_pad_atom(const float* __restrict__ src, float* __restrict__ dst) {
  int i = blockIdx.x * blockDim.x + threadIdx.x;
  if (i >= NN * ORIGP) return;
  int n = i / ORIGP, c = i - n * ORIGP;
  dst[i] = (c < ORIGF) ? src[n * ORIGF + c] : 0.f;
}

// ------------------- column stats (mean/var), deterministic ----------------
__global__ void k_colstats_part(const float* __restrict__ X, int Rr, int Cc, int chunk,
                                float* __restrict__ ps, float* __restrict__ ps2) {
  int c  = blockIdx.x * 64 + threadIdx.x;
  int r0 = blockIdx.y * chunk;
  int r1 = r0 + chunk; if (r1 > Rr) r1 = Rr;
  float s = 0.f, s2 = 0.f;
  for (int r = r0; r < r1; ++r) { float v = X[r * Cc + c]; s += v; s2 += v * v; }
  ps [blockIdx.y * Cc + c] = s;
  ps2[blockIdx.y * Cc + c] = s2;
}

__global__ void k_colstats_final(const float* __restrict__ ps, const float* __restrict__ ps2,
                                 int Rr, int Cc, int RS,
                                 float* __restrict__ mean, float* __restrict__ var) {
  int c = blockIdx.x * 64 + threadIdx.x;
  float s = 0.f, s2 = 0.f;
  for (int i = 0; i < RS; ++i) { s += ps[i * Cc + c]; s2 += ps2[i * Cc + c]; }
  float mu = s / (float)Rr;
  mean[c] = mu;
  var[c]  = s2 / (float)Rr - mu * mu;
}

// ---- conv epilogue: summed[n,d] = sum_m sigmoid(bn1(g[:,d]))*softplus(bn1(g[:,d+128]))
__global__ void k_conv_post(const float* __restrict__ g,
                            const float* __restrict__ mean, const float* __restrict__ var,
                            const float* __restrict__ g1, const float* __restrict__ b1,
                            float* __restrict__ summed) {
  int i = blockIdx.x * blockDim.x + threadIdx.x;
  if (i >= NN * DD) return;
  int n = i >> 7, d = i & 127;
  float r1 = rsqrtf(var[d] + EPSB), r2 = rsqrtf(var[d + 128] + EPSB);
  float m1 = mean[d], m2 = mean[d + 128];
  float ga = g1[d], ba = b1[d], gb = g1[d + 128], bb = b1[d + 128];
  float acc = 0.f;
#pragma unroll
  for (int m = 0; m < MM; ++m) {
    const float* row = g + (n * MM + m) * 256;
    float f = ga * (row[d] - m1) * r1 + ba;
    float c = gb * (row[d + 128] - m2) * r2 + bb;
    acc += sigmoidf_(f) * softplusf_(c);
  }
  summed[i] = acc;
}

__global__ void k_conv_finish(float* __restrict__ x, const float* __restrict__ summed,
                              const float* __restrict__ mean, const float* __restrict__ var,
                              const float* __restrict__ g2, const float* __restrict__ b2) {
  int i = blockIdx.x * blockDim.x + threadIdx.x;
  if (i >= NN * DD) return;
  int d = i & 127;
  float s = g2[d] * (summed[i] - mean[d]) * rsqrtf(var[d] + EPSB) + b2[d];
  x[i] = softplusf_(x[i] + s);
}

// ------------------------------ degrees ------------------------------------
__global__ void k_count(const int* __restrict__ nbr, int* __restrict__ indeg) {
  int i = blockIdx.x * blockDim.x + threadIdx.x;
  if (i < NN * MM) atomicAdd(&indeg[nbr[i]], 1);
}

__global__ void k_degapply(float* __restrict__ x, const int* __restrict__ indeg,
                           const float* __restrict__ zin, const float* __restrict__ zout) {
  int i = blockIdx.x * blockDim.x + threadIdx.x;
  if (i >= NN * DD) return;
  int n = i >> 7, d = i & 127;
  int dg = indeg[n]; if (dg > 19) dg = 19;
  x[i] += zin[dg * DD + d] + zout[12 * DD + d];
}

// ------------------------------ layernorm ----------------------------------
__global__ void k_layernorm(const float* __restrict__ X, const float* __restrict__ g,
                            const float* __restrict__ b, float* __restrict__ Y) {
  int n = blockIdx.x, d = threadIdx.x;
  float v = X[n * DD + d];
  __shared__ float s1[DD], s2[DD];
  s1[d] = v; s2[d] = v * v; __syncthreads();
  for (int off = 64; off > 0; off >>= 1) {
    if (d < off) { s1[d] += s1[d + off]; s2[d] += s2[d + off]; }
    __syncthreads();
  }
  float mu  = s1[0] / (float)DD;
  float var = s2[0] / (float)DD - mu * mu;
  Y[n * DD + d] = g[d] * (v - mu) * rsqrtf(var + EPSB) + b[d];
}

// ------------------- sparse masked attention (one wave per (n,h)) ----------
__global__ void k_attn(const float* __restrict__ q, const float* __restrict__ k,
                       const float* __restrict__ v, const int* __restrict__ nbr,
                       float* __restrict__ o) {
  int tid  = blockIdx.x * blockDim.x + threadIdx.x;
  int lane = tid & 31;
  int w    = tid >> 5;
  if (w >= NN * NHEAD) return;
  int n = w >> 2, hh = w & 3;

  int idxs[MM];
#pragma unroll
  for (int j = 0; j < MM; ++j) idxs[j] = nbr[n * MM + j];

  const float* qp = q + ((hh * NN) + n) * DD + lane * 4;
  float q0 = qp[0], q1 = qp[1], q2 = qp[2], q3 = qp[3];

  float s[MM];
  float mx = -1e30f;
#pragma unroll
  for (int j = 0; j < MM; ++j) {
    bool dup = false;
#pragma unroll
    for (int i2 = 0; i2 < MM; ++i2)
      if (i2 < j) dup = dup || (idxs[i2] == idxs[j]);
    if (dup) { s[j] = -1e30f; continue; }
    const float* kp = k + ((hh * NN) + idxs[j]) * DD + lane * 4;
    float part = q0 * kp[0] + q1 * kp[1] + q2 * kp[2] + q3 * kp[3];
    for (int off = 16; off > 0; off >>= 1) part += __shfl_xor(part, off, 32);
    s[j] = part * 0.08838834764831845f;   // 1/sqrt(128)
    mx = fmaxf(mx, s[j]);
  }
  float denom = 0.f, p[MM];
#pragma unroll
  for (int j = 0; j < MM; ++j) { p[j] = expf(s[j] - mx); denom += p[j]; }

  float a0 = 0.f, a1 = 0.f, a2 = 0.f, a3 = 0.f;
#pragma unroll
  for (int j = 0; j < MM; ++j) {
    if (p[j] == 0.f) continue;   // wave-uniform (p[j] identical across lanes)
    const float* vp = v + ((hh * NN) + idxs[j]) * DD + lane * 4;
    a0 += p[j] * vp[0]; a1 += p[j] * vp[1]; a2 += p[j] * vp[2]; a3 += p[j] * vp[3];
  }
  float inv = 1.f / denom;
  float* op = o + n * (NHEAD * DD) + hh * DD + lane * 4;
  op[0] = a0 * inv; op[1] = a1 * inv; op[2] = a2 * inv; op[3] = a3 * inv;
}

// ---------------------------------------------------------------------------
extern "C" void kernel_launch(void* const* d_in, const int* in_sizes, int n_in,
                              void* d_out, int out_size, void* d_ws, size_t ws_size,
                              hipStream_t stream) {
  (void)in_sizes; (void)n_in; (void)out_size; (void)ws_size;
  // Inputs flattened depth-first in setup_inputs() insertion order:
  // 0 atom_fea, 1 nbr_fea, 2 cond_emb,
  // 3 emb_W, 4 emb_b, [5+6c .. 10+6c] conv c: fc_W, fc_b, bn1_g, bn1_b, bn2_g, bn2_b,
  // 23 z_in, 24 z_out,
  // 25..40 glayer: Wq,bq,Wk,bk,Wv,bv,Wo,bo,ln1_g,ln1_b,ln2_g,ln2_b,ff_W1,ff_b1,ff_W2,ff_b2,
  // 41..44 cp_W1,cp_b1,cp_W2,cp_b2, 45 nbr_fea_idx (int32)
  const float* atom_fea = (const float*)d_in[0];
  const float* nbr_fea  = (const float*)d_in[1];
  const float* cond_emb = (const float*)d_in[2];
  const float* emb_W    = (const float*)d_in[3];
  const float* emb_b    = (const float*)d_in[4];
  const float* z_in     = (const float*)d_in[23];
  const float* z_out    = (const float*)d_in[24];
  const float* Wq = (const float*)d_in[25]; const float* bq = (const float*)d_in[26];
  const float* Wk = (const float*)d_in[27]; const float* bk = (const float*)d_in[28];
  const float* Wv = (const float*)d_in[29]; const float* bv = (const float*)d_in[30];
  const float* Wo = (const float*)d_in[31]; const float* bo = (const float*)d_in[32];
  const float* ln1g = (const float*)d_in[33]; const float* ln1b = (const float*)d_in[34];
  const float* ln2g = (const float*)d_in[35]; const float* ln2b = (const float*)d_in[36];
  const float* ffW1 = (const float*)d_in[37]; const float* ffb1 = (const float*)d_in[38];
  const float* ffW2 = (const float*)d_in[39]; const float* ffb2 = (const float*)d_in[40];
  const float* cpW1 = (const float*)d_in[41]; const float* cpb1 = (const float*)d_in[42];
  const float* cpW2 = (const float*)d_in[43]; const float* cpb2 = (const float*)d_in[44];
  const int* nbr_idx = (const int*)d_in[45];
  float* out = (float*)d_out;

  // ---- workspace carve-up (floats, 256B aligned chunks) ----
  float* wsf = reinterpret_cast<float*>(d_ws);
  size_t off = 0;
  auto alloc  = [&](size_t nf) { float* p = wsf + off; off += (nf + 63) & ~(size_t)63; return p; };
  auto halloc = [&](size_t nh) { return (_Float16*)alloc(nh / 2); };
  float* x      = alloc((size_t)NN * DD);
  float* tbuf   = alloc((size_t)NN * DD);
  float* xp     = alloc((size_t)NN * DD);
  float* ffbuf  = alloc((size_t)NN * DD);
  float* summed = alloc((size_t)NN * DD);
  float* apad   = alloc((size_t)NN * ORIGP);
  float* qb     = alloc((size_t)NHEAD * NN * DD);
  float* kb2    = alloc((size_t)NHEAD * NN * DD);
  float* vb     = alloc((size_t)NHEAD * NN * DD);
  float* ob     = alloc((size_t)NN * NHEAD * DD);
  float* gbuf   = alloc((size_t)NN * MM * 256);
  float* psum   = alloc(64 * 256);
  float* psumsq = alloc(64 * 256);
  float* meanv  = alloc(256);
  float* varv   = alloc(256);
  int*   indeg  = (int*)alloc(NN);
  // swizzled f16 weights
  _Float16* sw_emb = halloc((size_t)ORIGP * DD);
  _Float16* sw_cp1 = halloc((size_t)DD * DD);
  _Float16* sw_cp2 = halloc((size_t)DD * DD);
  _Float16* sw_q   = halloc((size_t)NHEAD * DD * DD);
  _Float16* sw_k   = halloc((size_t)NHEAD * DD * DD);
  _Float16* sw_v   = halloc((size_t)NHEAD * DD * DD);
  _Float16* sw_wo  = halloc((size_t)NHEAD * DD * DD);
  _Float16* sw_ff1 = halloc((size_t)DD * DD);
  _Float16* sw_ff2 = halloc((size_t)DD * DD);
  _Float16* sw_fc0 = halloc((size_t)320 * 256);
  _Float16* sw_fc1 = halloc((size_t)320 * 256);
  _Float16* sw_fc2 = halloc((size_t)320 * 256);
  _Float16* sw_fc[3] = {sw_fc0, sw_fc1, sw_fc2};

  auto swz = [&](const float* B, _Float16* Bs, int K, int NC) {
    k_swz_b<<<(K * NC + 255) / 256, 256, 0, stream>>>(B, Bs, K, NC);
  };
  auto colstats = [&](const float* Xm, int Rr, int Cc) {
    const int RS = 64;
    int chunk = (Rr + RS - 1) / RS;
    k_colstats_part<<<dim3(Cc / 64, RS), 64, 0, stream>>>(Xm, Rr, Cc, chunk, psum, psumsq);
    k_colstats_final<<<Cc / 64, 64, 0, stream>>>(psum, psumsq, Rr, Cc, RS, meanv, varv);
  };

  // 0) swizzle all weights once
  swz(emb_W, sw_emb, ORIGP, DD);   // rows >= 92 multiply zero-padded A columns
  swz(cpW1, sw_cp1, DD, DD);
  swz(cpW2, sw_cp2, DD, DD);
  for (int h = 0; h < NHEAD; ++h) {
    swz(Wq + h * DD * DD, sw_q + (size_t)h * DD * DD, DD, DD);
    swz(Wk + h * DD * DD, sw_k + (size_t)h * DD * DD, DD, DD);
    swz(Wv + h * DD * DD, sw_v + (size_t)h * DD * DD, DD, DD);
  }
  swz(Wo, sw_wo, NHEAD * DD, DD);
  swz(ffW1, sw_ff1, DD, DD);
  swz(ffW2, sw_ff2, DD, DD);
  for (int c = 0; c < 3; ++c) swz((const float*)d_in[5 + c * 6], sw_fc[c], 320, 256);

  // 1) embedding: x = atom_fea @ emb_W + emb_b   (A zero-padded to K=96)
  k_pad_atom<<<(NN * ORIGP + 255) / 256, 256, 0, stream>>>(atom_fea, apad);
  launch_gemm<ORIGP, 0, false>(apad, sw_emb, emb_b, nullptr, x, NN, stream);

  // 2) conv layers
  for (int c = 0; c < 3; ++c) {
    const float* fcb  = (const float*)d_in[5 + c * 6 + 1];
    const float* bn1g = (const float*)d_in[5 + c * 6 + 2];
    const float* bn1b = (const float*)d_in[5 + c * 6 + 3];
    const float* bn2g = (const float*)d_in[5 + c * 6 + 4];
    const float* bn2b = (const float*)d_in[5 + c * 6 + 5];
    int waves = (NN * MM / 16) * 4;               // 12288 waves (16x64 each)
    k_conv_gemm<<<waves / 4, 128, 0, stream>>>(x, nbr_fea, nbr_idx, sw_fc[c], fcb, gbuf);
    colstats(gbuf, NN * MM, 256);
    k_conv_post<<<(NN * DD) / 256, 256, 0, stream>>>(gbuf, meanv, varv, bn1g, bn1b, summed);
    colstats(summed, NN, DD);
    k_conv_finish<<<(NN * DD) / 256, 256, 0, stream>>>(x, summed, meanv, varv, bn2g, bn2b);
  }

  // 3) conditional projection: x += silu(cond @ W1 + b1) @ W2 + b2
  launch_gemm<DD, 2, false>(cond_emb, sw_cp1, cpb1, nullptr, tbuf, NN, stream);
  launch_gemm<DD, 0, true >(tbuf, sw_cp2, cpb2, x, x, NN, stream);

  // 4) degree embeddings
  hipMemsetAsync(indeg, 0, NN * sizeof(int), stream);
  k_count<<<(NN * MM + 255) / 256, 256, 0, stream>>>(nbr_idx, indeg);
  k_degapply<<<(NN * DD) / 256, 256, 0, stream>>>(x, indeg, z_in, z_out);

  // 5) graph transformer layer
  k_layernorm<<<NN, DD, 0, stream>>>(x, ln1g, ln1b, tbuf);
  for (int h = 0; h < NHEAD; ++h) {
    launch_gemm<DD, 0, false>(tbuf, sw_q + (size_t)h * DD * DD, bq + h * DD, nullptr,
                              qb  + (size_t)h * NN * DD, NN, stream);
    launch_gemm<DD, 0, false>(tbuf, sw_k + (size_t)h * DD * DD, bk + h * DD, nullptr,
                              kb2 + (size_t)h * NN * DD, NN, stream);
    launch_gemm<DD, 0, false>(tbuf, sw_v + (size_t)h * DD * DD, bv + h * DD, nullptr,
                              vb  + (size_t)h * NN * DD, NN, stream);
  }
  k_attn<<<(NN * NHEAD * 32) / 256, 256, 0, stream>>>(qb, kb2, vb, nbr_idx, ob);
  launch_gemm<NHEAD * DD, 0, true>(ob, sw_wo, bo, x, xp, NN, stream);
  k_layernorm<<<NN, DD, 0, stream>>>(xp, ln2g, ln2b, tbuf);
  launch_gemm<DD, 1, false>(tbuf, sw_ff1, ffb1, nullptr, ffbuf, NN, stream);
  launch_gemm<DD, 0, true >(ffbuf, sw_ff2, ffb2, xp, out, NN, stream);
}
